// MultiHeadedAttention_3_type_min_18631568130080
// MI455X (gfx1250) — matrix operand order repaired and checked by reading, not compiled, not run on hardware
//
#include <hip/hip_runtime.h>

// ---------------------------------------------------------------------------
// Shapes (fixed by the reference): B=4, N=256, D=256, H=8
//   ft   : [B*N*N, 512]   (query || key, both row-contiguous with D=256)
//   W    : [512, 8], bias : [8]
//   attn : [B*N*N, 8] = sigmoid(ft @ W + b)        (WMMA bf16 GEMM, HBM-bound)
//   mask : union of top-k indices over dim-2 slices (radix select, wave32)
//   out  : attn * (roi1+roi2) * mask[m]
// ---------------------------------------------------------------------------

typedef __attribute__((ext_vector_type(16))) __bf16 v16bf;
typedef __attribute__((ext_vector_type(8)))  float  v8f;

#define NB        4
#define NN        256
#define DM        256
#define HEADS     8
#define KDIM      512
#define ROWS      (NB * NN * NN)          // 262144 GEMM rows
#define TILES     (ROWS / 16)             // 16384 row-tiles of 16
#define ATTN_ELEMS (ROWS * HEADS)         // 2097152 floats (8 MB)
#define WT_STRIDE 528                     // padded LDS row stride (bf16 elems)

// ---------------------------------------------------------------------------
// Kernel 1: attn = sigmoid(ft @ W + b) via v_wmma_f32_16x16x32_bf16
// One wave per 16-row tile; K=512 -> 16 WMMA ops. W^T staged in LDS as bf16,
// padded to 16 columns (cols 8..15 zero).
// ---------------------------------------------------------------------------
__global__ void attn_gemm_wmma(const float* __restrict__ q,
                               const float* __restrict__ kk,
                               const float* __restrict__ W,
                               const float* __restrict__ bias,
                               float* __restrict__ attn)
{
    __shared__ __align__(32) __bf16 Wt[16 * WT_STRIDE];

    const int tid = threadIdx.x;
    // Stage W^T (16 x 512 bf16, padded) into LDS
    for (int i = tid; i < 16 * KDIM; i += blockDim.x) {
        const int col = i >> 9;          // / 512
        const int kd  = i & 511;
        const float v = (col < HEADS) ? W[kd * HEADS + col] : 0.0f;
        Wt[col * WT_STRIDE + kd] = (__bf16)v;
    }
    __syncthreads();

    const int wave = tid >> 5;
    const int lane = tid & 31;
    const int r    = lane & 15;          // A-matrix row M handled by this lane
    const int half = lane >> 4;          // K-half selector (this code's contiguous k = 16*half + i)
    const int tile = blockIdx.x * 8 + wave;
    const long Row = (long)tile * 16 + r;

    // Prefetch the key row while query chunks stream first
    __builtin_prefetch(kk + Row * DM, 0, 1);

    v8f acc = {};
    #pragma unroll
    for (int c = 0; c < 16; ++c) {
        // A fragment: 16x32 bf16. lane holds row M=lane&15;
        // elems 0..15 -> K = c*32 + half*16 + 0..15
        const float* src = (c < 8) ? (q  + Row * DM + c * 32)
                                   : (kk + Row * DM + (c - 8) * 32);
        const float4 f0 = *(const float4*)(src + half * 16);
        const float4 f1 = *(const float4*)(src + half * 16 + 4);
        const float4 f2 = *(const float4*)(src + half * 16 + 8);
        const float4 f3 = *(const float4*)(src + half * 16 + 12);

        v16bf a;
        a[0]  = (__bf16)f0.x; a[1]  = (__bf16)f0.y;
        a[2]  = (__bf16)f0.z; a[3]  = (__bf16)f0.w;
        a[4]  = (__bf16)f1.x; a[5]  = (__bf16)f1.y;
        a[6]  = (__bf16)f1.z; a[7]  = (__bf16)f1.w;
        a[8]  = (__bf16)f2.x; a[9]  = (__bf16)f2.y;
        a[10] = (__bf16)f2.z; a[11] = (__bf16)f2.w;
        a[12] = (__bf16)f3.x; a[13] = (__bf16)f3.y;
        a[14] = (__bf16)f3.z; a[15] = (__bf16)f3.w;

        // B fragment: 32x16 bf16. lane holds col N=lane&15, elem i -> K = c*32 + half*16 + i
        const v16bf bm = *(const v16bf*)(Wt + (lane & 15) * WT_STRIDE + c * 32 + half * 16);

        acc = __builtin_amdgcn_wmma_f32_16x16x32_bf16(
                  false, a, false, bm, (short)0, acc, false, false);
    }

    // D layout: lane holds col N=lane&15; VGPR rr -> row M = rr + 8*half
    const int  col = lane & 15;
    const long R0  = (long)tile * 16;
    if (col < HEADS) {
        const float bv = bias[col];
        #pragma unroll
        for (int rr = 0; rr < 8; ++rr) {
            const int  M = rr + 8 * half;
            const float x = acc[rr] + bv;
            const float s = 1.0f / (1.0f + __expf(-x));
            attn[(R0 + M) * HEADS + col] = s;
        }
    }
}

// ---------------------------------------------------------------------------
// Kernel 2: union-of-top-k mask along dim 2.
// One wave per (b,n,h) slice of 256 values; 32-step radix select for the
// k-th largest, wave32 shuffle reduction for the counts; mark >= threshold.
// ---------------------------------------------------------------------------
__device__ __forceinline__ unsigned mapf(float x) {
    const unsigned u = __float_as_uint(x);
    return (u & 0x80000000u) ? ~u : (u | 0x80000000u);   // order-preserving
}

__device__ __forceinline__ unsigned kth_thresh(const unsigned* u, int k) {
    unsigned p = 0u;
    for (int b = 31; b >= 0; --b) {
        const unsigned cand = p | (1u << b);
        int c = 0;
        #pragma unroll
        for (int j = 0; j < 8; ++j)
            c += ((u[j] >> b) >= (cand >> b)) ? 1 : 0;
        #pragma unroll
        for (int off = 16; off > 0; off >>= 1)
            c += __shfl_xor(c, off, 32);
        if (c >= k) p = cand;
    }
    return p;   // exact k-th largest key
}

__global__ void topk_union_mask(const float* __restrict__ attn,
                                const float* __restrict__ roi1,
                                const float* __restrict__ roi2,
                                float* __restrict__ maskf)
{
    const int wave = threadIdx.x >> 5;
    const int lane = threadIdx.x & 31;
    const int s    = blockIdx.x * 8 + wave;   // 0..8191 slices
    const int bn   = s >> 3;                  // (b*N+n)
    const int h    = s & 7;

    unsigned u1[8], u2[8], n1[8], n2[8];
    int mlist[8];
    #pragma unroll
    for (int j = 0; j < 8; ++j) {
        const int m = lane + 32 * j;
        mlist[j] = m;
        const float av = attn[((long)bn * NN + m) * HEADS + h];
        const float r1 = roi1[bn * NN + m];
        const float r2 = roi2[bn * NN + m];
        u1[j] = mapf(av * r1);
        u2[j] = mapf(av * r2);
        n1[j] = ~u1[j];                       // bottom-k == top-k of complement
        n2[j] = ~u2[j];
    }

    unsigned t;
    t = kth_thresh(u1, 64);
    #pragma unroll
    for (int j = 0; j < 8; ++j) if (u1[j] >= t) maskf[mlist[j]] = 1.0f;
    t = kth_thresh(n1, 64);
    #pragma unroll
    for (int j = 0; j < 8; ++j) if (n1[j] >= t) maskf[mlist[j]] = 1.0f;
    t = kth_thresh(u2, 128);
    #pragma unroll
    for (int j = 0; j < 8; ++j) if (u2[j] >= t) maskf[mlist[j]] = 1.0f;
    t = kth_thresh(n2, 64);
    #pragma unroll
    for (int j = 0; j < 8; ++j) if (n2[j] >= t) maskf[mlist[j]] = 1.0f;
}

// ---------------------------------------------------------------------------
// Kernel 3: out = attn * (roi1 + roi2) * mask[m], float4-vectorized.
// ---------------------------------------------------------------------------
__global__ void finalize_out(const float* __restrict__ attn,
                             const float* __restrict__ roi1,
                             const float* __restrict__ roi2,
                             const float* __restrict__ maskf,
                             float* __restrict__ out)
{
    const long i4  = ((long)blockIdx.x * blockDim.x + threadIdx.x) * 4;
    const long bnm = i4 >> 3;                 // (b*N+n)*N + m
    const int  m   = (int)(bnm & (NN - 1));
    const float sc = (roi1[bnm] + roi2[bnm]) * maskf[m];
    const float4 a = *(const float4*)(attn + i4);
    float4 o;
    o.x = a.x * sc; o.y = a.y * sc; o.z = a.z * sc; o.w = a.w * sc;
    *(float4*)(out + i4) = o;
}

// ---------------------------------------------------------------------------
extern "C" void kernel_launch(void* const* d_in, const int* in_sizes, int n_in,
                              void* d_out, int out_size, void* d_ws, size_t ws_size,
                              hipStream_t stream)
{
    const float* q    = (const float*)d_in[0];   // query  [4,256,256,256]
    const float* k    = (const float*)d_in[1];   // key    [4,256,256,256]
    const float* roi1 = (const float*)d_in[2];   // [4,256,256]
    const float* roi2 = (const float*)d_in[3];   // [4,256,256]
    const float* W    = (const float*)d_in[4];   // [512,8]
    const float* bias = (const float*)d_in[5];   // [8]
    (void)in_sizes; (void)n_in; (void)out_size; (void)ws_size;

    float* attn  = (float*)d_ws;                 // 8 MB scratch
    float* maskf = attn + ATTN_ELEMS;            // 256 floats

    hipMemsetAsync(maskf, 0, NN * sizeof(float), stream);

    attn_gemm_wmma<<<TILES / 8, 256, 0, stream>>>(q, k, W, bias, attn);
    topk_union_mask<<<8192 / 8, 256, 0, stream>>>(attn, roi1, roi2, maskf);
    finalize_out<<<(ATTN_ELEMS / 4) / 256, 256, 0, stream>>>(attn, roi1, roi2, maskf,
                                                             (float*)d_out);
}
